// WriteMemory_15453292331557
// MI455X (gfx1250) — matrix operand order repaired and checked by reading, not compiled, not run on hardware
//
#include <hip/hip_runtime.h>
#include <hip/hip_bf16.h>

typedef __attribute__((ext_vector_type(16))) _Float16 v16h;
typedef __attribute__((ext_vector_type(8)))  float    v8f;
typedef _Float16 half_t;

#define BATCH  512
#define DMODEL 2048
#define NHEAD  8
#define DMEM   128
#define MMEM   512
#define MAX_WRITE_FACTOR 0.7f

// elu(x)+1
__device__ __forceinline__ float actf(float x) { return x > 0.f ? x + 1.f : __expf(x); }
__device__ __forceinline__ float sigm(float x) { return 1.f / (1.f + __expf(-x)); }

// ---- WMMA fragment gathers from LDS (f16, row-major, 16B-aligned rows) ----
// A operand (16x32): lane m = l&15, hi = l>>4; element e -> K = e + (e>=8?8:0) + 8*hi
__device__ __forceinline__ v16h frag_a(const half_t* row, int hi) {
  v16h f;
  ((float4*)&f)[0] = *(const float4*)(row + hi * 8);
  ((float4*)&f)[1] = *(const float4*)(row + hi * 8 + 16);
  return f;
}
// B operand (32x16), stored n-major in LDS: lane n = l&15, hi = l>>4; element e -> K = e + (e>=8?8:0) + 8*hi
__device__ __forceinline__ v16h frag_b(const half_t* row, int hi) {
  v16h f;
  ((float4*)&f)[0] = *(const float4*)(row + hi * 8);
  ((float4*)&f)[1] = *(const float4*)(row + hi * 8 + 16);
  return f;
}

#define WMMA_F16(A, B, C) \
  __builtin_amdgcn_wmma_f32_16x16x32_f16(false, (A), false, (B), (short)0, (C), false, false)

// =====================================================================
// Kernel 1: C[512 x N] = act?( X[512 x 2048] * W[2048 x N] + bias ), f16 out
// grid: (N/128, 512/128), block 256 (8 waves: 2 m x 4 n), wave tile 64x32
// =====================================================================
template <bool DO_ACT>
__global__ __launch_bounds__(256) void proj_kernel(const float* __restrict__ X,
                                                   const float* __restrict__ W,
                                                   const float* __restrict__ bias,
                                                   half_t* __restrict__ Out, int N) {
  __shared__ half_t lA[128 * 48];  // X tile, row m x 32 k
  __shared__ half_t lB[128 * 48];  // W tile transposed: row n x 32 k
  const int t = threadIdx.x;
  const int m0 = blockIdx.y * 128, n0 = blockIdx.x * 128;
  const int lane = t & 31, w = t >> 5;
  const int ln = lane & 15, hi = lane >> 4;
  const int wm = w >> 2, wn = w & 3;

  v8f acc[4][2] = {};

  const int amm = t >> 1, akk = (t & 1) * 16;   // A loader: row, col base
  const int bkk = t >> 3, bnn = (t & 7) * 16;   // B loader: k row, n col base

  for (int k0 = 0; k0 < DMODEL; k0 += 32) {
    const float* xa = X + (size_t)(m0 + amm) * DMODEL + k0 + akk;
#pragma unroll
    for (int j = 0; j < 16; ++j) lA[amm * 48 + akk + j] = (half_t)xa[j];
    const float* xw = W + (size_t)(k0 + bkk) * N + n0 + bnn;
#pragma unroll
    for (int j = 0; j < 16; ++j) lB[(bnn + j) * 48 + bkk] = (half_t)xw[j];
    __syncthreads();

    v16h af[4], bf[2];
#pragma unroll
    for (int i = 0; i < 4; ++i) af[i] = frag_a(&lA[(wm * 64 + i * 16 + ln) * 48], hi);
#pragma unroll
    for (int j = 0; j < 2; ++j) bf[j] = frag_b(&lB[(wn * 32 + j * 16 + ln) * 48], hi);
#pragma unroll
    for (int i = 0; i < 4; ++i)
#pragma unroll
      for (int j = 0; j < 2; ++j) acc[i][j] = WMMA_F16(af[i], bf[j], acc[i][j]);
    __syncthreads();
  }

#pragma unroll
  for (int i = 0; i < 4; ++i)
#pragma unroll
    for (int j = 0; j < 2; ++j) {
      const int gn = n0 + wn * 32 + j * 16 + ln;
      const float bv = bias[gn];
#pragma unroll
      for (int r = 0; r < 8; ++r) {
        const int gm = m0 + wm * 64 + i * 16 + r + hi * 8;
        float x = acc[i][j][r] + bv;
        if (DO_ACT) x = actf(x);
        Out[(size_t)gm * N + gn] = (half_t)x;
      }
    }
}

// =====================================================================
// Kernel 2: aq[m][d] = act(addresses[m][d]) -> f16
// =====================================================================
__global__ __launch_bounds__(256) void aq_kernel(const float* __restrict__ addr,
                                                 half_t* __restrict__ aq) {
  const int i = blockIdx.x * 256 + threadIdx.x;
  if (i < MMEM * DMEM) aq[i] = (half_t)actf(addr[i]);
}

// =====================================================================
// Kernel 3: per batch b: matT[b][e][d] = sum_h ak[b,h,d]*v[b,h,e] (f16),
//           norm[b][d] = sum_h ak[b,h,d] (f32).   One WG per batch.
// =====================================================================
__global__ __launch_bounds__(256) void outer_kernel(const half_t* __restrict__ ak,
                                                    const half_t* __restrict__ vp,
                                                    half_t* __restrict__ matT,
                                                    float* __restrict__ norm) {
  __shared__ float sak[NHEAD * DMEM];       // 8x128
  __shared__ float sv[NHEAD * 2 * DMEM];    // 8x256
  const int b = blockIdx.x, t = threadIdx.x;
  for (int i = t; i < NHEAD * DMEM; i += 256) sak[i] = (float)ak[(size_t)b * (NHEAD * DMEM) + i];
  for (int i = t; i < NHEAD * 2 * DMEM; i += 256) sv[i] = (float)vp[(size_t)b * (NHEAD * 2 * DMEM) + i];
  __syncthreads();

  const int d = t & 127;
  const int e0 = (t >> 7) * 128;
  float a[NHEAD];
#pragma unroll
  for (int h = 0; h < NHEAD; ++h) a[h] = sak[h * DMEM + d];
  if (t < DMEM) {
    float s = 0.f;
#pragma unroll
    for (int h = 0; h < NHEAD; ++h) s += a[h];
    norm[(size_t)b * DMEM + d] = s;
  }
  half_t* mt = matT + (size_t)b * (2 * DMEM * DMEM);
  for (int e = e0; e < e0 + 128; ++e) {
    float s = 0.f;
#pragma unroll
    for (int h = 0; h < NHEAD; ++h) s += a[h] * sv[h * 256 + e];
    mt[(size_t)e * DMEM + d] = (half_t)s;
  }
}

// =====================================================================
// Kernel 4: den[b][m] = dot(norm[b,:], aq[m,:]) + 1e-5.  512x512, K=128.
// grid (4,4), block 256, wave tile 64x32.
// =====================================================================
__global__ __launch_bounds__(256) void denom_kernel(const float* __restrict__ norm,
                                                    const half_t* __restrict__ aq,
                                                    float* __restrict__ den) {
  __shared__ half_t lA[128 * 48];  // norm rows (b) x k
  __shared__ half_t lB[128 * 48];  // aq rows (m) x k
  const int t = threadIdx.x;
  const int b0 = blockIdx.x * 128, m0 = blockIdx.y * 128;
  const int lane = t & 31, w = t >> 5, ln = lane & 15, hi = lane >> 4;
  const int wm = w >> 2, wn = w & 3;
  v8f acc[4][2] = {};
  const int rr = t >> 1, kk = (t & 1) * 16;

  for (int k0 = 0; k0 < DMEM; k0 += 32) {
    const float* na = norm + (size_t)(b0 + rr) * DMEM + k0 + kk;
#pragma unroll
    for (int j = 0; j < 16; ++j) lA[rr * 48 + kk + j] = (half_t)na[j];
    const half_t* qa = aq + (size_t)(m0 + rr) * DMEM + k0 + kk;
    *(float4*)&lB[rr * 48 + kk]     = *(const float4*)qa;
    *(float4*)&lB[rr * 48 + kk + 8] = *(const float4*)(qa + 8);
    __syncthreads();

    v16h af[4], bf[2];
#pragma unroll
    for (int i = 0; i < 4; ++i) af[i] = frag_a(&lA[(wm * 64 + i * 16 + ln) * 48], hi);
#pragma unroll
    for (int j = 0; j < 2; ++j) bf[j] = frag_b(&lB[(wn * 32 + j * 16 + ln) * 48], hi);
#pragma unroll
    for (int i = 0; i < 4; ++i)
#pragma unroll
      for (int j = 0; j < 2; ++j) acc[i][j] = WMMA_F16(af[i], bf[j], acc[i][j]);
    __syncthreads();
  }
#pragma unroll
  for (int i = 0; i < 4; ++i)
#pragma unroll
    for (int j = 0; j < 2; ++j)
#pragma unroll
      for (int r = 0; r < 8; ++r)
        den[(size_t)(b0 + wm * 64 + i * 16 + r + hi * 8) * MMEM + m0 + wn * 32 + j * 16 + ln] =
            acc[i][j][r] + 1e-5f;
}

// =====================================================================
// Kernel 5 (fused main): per (b, m-tile of 128):
//   num = aq_tile(128x128) @ matT[b]ᵀ -> 128x256 (upd | logits), WMMA
//   then elementwise memory update, reading memories/write_mass once.
// Wave pairing: wave owns update cols [d0,d0+32) AND logit cols [128+d0,...),
// so upd and its logit land in the same lane/VGPR slot -> register epilogue.
// grid (4, 512), block 256.
// =====================================================================
__global__ __launch_bounds__(256) void fuse_kernel(const half_t* __restrict__ aq,
                                                   const half_t* __restrict__ matT,
                                                   const float* __restrict__ den,
                                                   const float* __restrict__ memories,
                                                   const float* __restrict__ wmass,
                                                   const float* __restrict__ interp,
                                                   const unsigned char* __restrict__ bmask,
                                                   float* __restrict__ out_mem,
                                                   float* __restrict__ out_mass) {
  __shared__ half_t sA[128 * 48];  // aq rows (m) x k-chunk
  __shared__ half_t sB[256 * 48];  // matT rows (e=0..255) x k-chunk
  __shared__ float sDen[128];
  const int t = threadIdx.x;
  const int b = blockIdx.y, m0 = blockIdx.x * 128;
  const int lane = t & 31, w = t >> 5, ln = lane & 15, hi = lane >> 4;
  const int wm = w >> 2, we = w & 3;

  if (t < 128) sDen[t] = den[(size_t)b * MMEM + m0 + t];

  v8f accU[4][2] = {};
  v8f accL[4][2] = {};
  const int mm = t >> 1, kk = (t & 1) * 16;
  const half_t* mtb = matT + (size_t)b * (2 * DMEM * DMEM);

  for (int k0 = 0; k0 < DMEM; k0 += 32) {
    const half_t* qa = aq + (size_t)(m0 + mm) * DMEM + k0 + kk;
    *(float4*)&sA[mm * 48 + kk]     = *(const float4*)qa;
    *(float4*)&sA[mm * 48 + kk + 8] = *(const float4*)(qa + 8);
    const half_t* mb = mtb + (size_t)t * DMEM + k0;  // row e = t (0..255)
#pragma unroll
    for (int q = 0; q < 4; ++q) *(float4*)&sB[t * 48 + q * 8] = *(const float4*)(mb + q * 8);
    __syncthreads();

    v16h af[4], bu[2], bl[2];
#pragma unroll
    for (int i = 0; i < 4; ++i) af[i] = frag_a(&sA[(wm * 64 + i * 16 + ln) * 48], hi);
#pragma unroll
    for (int j = 0; j < 2; ++j) {
      bu[j] = frag_b(&sB[(we * 32 + j * 16 + ln) * 48], hi);
      bl[j] = frag_b(&sB[(128 + we * 32 + j * 16 + ln) * 48], hi);
    }
#pragma unroll
    for (int i = 0; i < 4; ++i)
#pragma unroll
      for (int j = 0; j < 2; ++j) {
        accU[i][j] = WMMA_F16(af[i], bu[j], accU[i][j]);
        accL[i][j] = WMMA_F16(af[i], bl[j], accL[i][j]);
      }
    __syncthreads();
  }

  const float wfscale = bmask[b] ? 0.0f : MAX_WRITE_FACTOR;
#pragma unroll
  for (int i = 0; i < 4; ++i)
#pragma unroll
    for (int j = 0; j < 2; ++j) {
      const int d = we * 32 + j * 16 + ln;
#pragma unroll
      for (int r = 0; r < 8; ++r) {
        const int m = wm * 64 + i * 16 + r + hi * 8;  // 0..127 within tile
        const float dn = sDen[m];
        const float u  = accU[i][j][r] / dn;
        const float wl = accL[i][j][r] / dn;
        const float wf = wfscale * sigm(wl) * sigm(interp[(size_t)(m0 + m) * DMEM + d]);
        const size_t idx = ((size_t)b * MMEM + m0 + m) * DMEM + d;
        const float mv = memories[idx];
        out_mem[idx]  = mv + (u - mv) * wf;
        out_mass[idx] = wmass[idx] + wf;
      }
    }
}

// =====================================================================
// Host side
// =====================================================================
static const size_t OFF_AK   = 0;                                            // 512*1024 f16 = 1 MB
static const size_t OFF_V    = OFF_AK + (size_t)BATCH * 1024 * 2;            // 512*2048 f16 = 2 MB
static const size_t OFF_AQ   = OFF_V + (size_t)BATCH * 2048 * 2;             // 512*128 f16
static const size_t OFF_NORM = OFF_AQ + (size_t)MMEM * DMEM * 2;             // 512*128 f32
static const size_t OFF_DEN  = OFF_NORM + (size_t)BATCH * DMEM * 4;          // 512*512 f32
static const size_t OFF_MT   = OFF_DEN + (size_t)BATCH * MMEM * 4;           // 512*256*128 f16 = 32 MB

extern "C" void kernel_launch(void* const* d_in, const int* in_sizes, int n_in,
                              void* d_out, int out_size, void* d_ws, size_t ws_size,
                              hipStream_t stream) {
  const float* key        = (const float*)d_in[0];
  const float* values     = (const float*)d_in[1];
  const float* memories   = (const float*)d_in[2];
  const float* write_mass = (const float*)d_in[3];
  const unsigned char* bmask = (const unsigned char*)d_in[4];
  const float* W_key = (const float*)d_in[5];
  const float* b_key = (const float*)d_in[6];
  const float* W_val = (const float*)d_in[7];
  const float* b_val = (const float*)d_in[8];
  const float* addresses = (const float*)d_in[9];
  const float* interp    = (const float*)d_in[10];

  char* ws = (char*)d_ws;
  half_t* ak   = (half_t*)(ws + OFF_AK);
  half_t* vp   = (half_t*)(ws + OFF_V);
  half_t* aq   = (half_t*)(ws + OFF_AQ);
  float*  norm = (float*)(ws + OFF_NORM);
  float*  den  = (float*)(ws + OFF_DEN);
  half_t* matT = (half_t*)(ws + OFF_MT);

  float* out_mem  = (float*)d_out;
  float* out_mass = out_mem + (size_t)BATCH * MMEM * DMEM;

  // 1) projections (ak = act(key W_key + b_key), v = values W_val + b_val)
  proj_kernel<true ><<<dim3(1024 / 128, BATCH / 128), 256, 0, stream>>>(key, W_key, b_key, ak, 1024);
  proj_kernel<false><<<dim3(2048 / 128, BATCH / 128), 256, 0, stream>>>(values, W_val, b_val, vp, 2048);
  // 2) aq = act(addresses)
  aq_kernel<<<(MMEM * DMEM + 255) / 256, 256, 0, stream>>>(addresses, aq);
  // 3) per-batch matrixᵀ + normalizer
  outer_kernel<<<BATCH, 256, 0, stream>>>(ak, vp, matT, norm);
  // 4) denominator
  denom_kernel<<<dim3(BATCH / 128, MMEM / 128), 256, 0, stream>>>(norm, aq, den);
  // 5) fused numerator GEMM + sigmoid/interpolate/mask + memory update
  fuse_kernel<<<dim3(MMEM / 128, BATCH), 256, 0, stream>>>(aq, matT, den, memories, write_mass,
                                                           interp, bmask, out_mem, out_mass);
}
